// GraniteMoeSharedDecoderLayer_46445776339257
// MI455X (gfx1250) — matrix operand order repaired and checked by reading, not compiled, not run on hardware
//
#include <hip/hip_runtime.h>

#define T_TOK   4096
#define DMODEL  1536
#define NH      24
#define NKV     8
#define HDIM    64
#define NEXP    32
#define TOPK_N  8
#define FEXP    512
#define FSHARED 1024
#define AMULT_F 0.015625f
#define RMULT_F 0.22f
#define RMS_EPS 1e-6f

typedef __bf16 bf16_t;
typedef __attribute__((ext_vector_type(16))) __bf16 v16bf;
typedef __attribute__((ext_vector_type(8)))  float  v8f;
typedef __attribute__((ext_vector_type(4)))  unsigned int v4u;
typedef __attribute__((ext_vector_type(8)))  int v8i;
typedef __attribute__((ext_vector_type(4)))  int v4i;

// ---------------------------------------------------------------------------
// WMMA helpers (CDNA5 16x16x32 bf16, f32 accumulate; wave32 fragment layouts
// per CDNA5 ISA 7.12.2)
// ---------------------------------------------------------------------------
__device__ __forceinline__ v8f wmma_bf16(v16bf a, v16bf b, v8f c) {
  return __builtin_amdgcn_wmma_f32_16x16x32_bf16(false, a, false, b, (short)0, c,
                                                 false, false);
}

// 16-bit A-matrix 16x32: VGPR i holds K pair; lanes 16-31 shifted in K.
__device__ __forceinline__ int kmap8(int i, int half) {
  return 2 * i + 16 * half;
}

// A fragment: this lane's matrix row data starts at Arow (contiguous K).
__device__ __forceinline__ v16bf load_a_frag(const bf16_t* __restrict__ Arow,
                                             int k0, int half) {
  v16bf a;
#pragma unroll
  for (int i = 0; i < 8; ++i) {
    int kb = k0 + kmap8(i, half);
    a[2 * i]     = Arow[kb];
    a[2 * i + 1] = Arow[kb + 1];
  }
  return a;
}

// B fragment (K x N, 32x16) straight from row-major f32 global memory,
// converted to bf16 in-register (used by the MoE kernels).
__device__ __forceinline__ v16bf load_b_frag_f32(const float* __restrict__ B,
                                                 int ldb, int k0, int n, int half) {
  v16bf b;
  const float* __restrict__ Bp = B + (size_t)(k0 + 16 * half) * ldb + n;
#pragma unroll
  for (int e = 0; e < 16; ++e) b[e] = (bf16_t)Bp[(size_t)e * ldb];
  return b;
}

// B fragment from an LDS-staged 32x64 f32 tile (TDM destination).
__device__ __forceinline__ v16bf load_b_frag_lds(const float* Bs, int nloc, int half) {
  v16bf b;
  const float* p = Bs + (16 * half) * 64 + nloc;
#pragma unroll
  for (int e = 0; e < 16; ++e) b[e] = (bf16_t)p[e * 64];
  return b;
}

// ---------------------------------------------------------------------------
// Tensor Data Mover: 2D f32 tile (tile_w x tile_h) from global -> LDS.
// D# bit layout per CDNA5 ISA:
//   group0: count=1 | lds_addr | global_addr[56:0] | type=2
//   group1: data_size=4B, tensor_dim0/1, tile_dim0/1, tensor_dim0_stride
// Issued by one wave; tracked by TENSORcnt. This toolchain exposes the
// 6-argument builtin: (v4u g0, v8i g1, v4i g2, v4i g3, v8i extra, i32 cpol).
// ---------------------------------------------------------------------------
__device__ __forceinline__ void tdm_load_2d_f32(unsigned lds_off, const float* gaddr,
                                                unsigned tile_w, unsigned tile_h,
                                                unsigned row_stride_elems) {
  unsigned long long ga = (unsigned long long)(const void*)gaddr;
  v4u g0;
  g0[0] = 1u;                                   // count=1 valid descriptor
  g0[1] = lds_off;                              // LDS byte address
  g0[2] = (unsigned)(ga & 0xffffffffu);         // global_addr[31:0]
  g0[3] = (unsigned)((ga >> 32) & 0x01ffffffu) | (2u << 30);  // [56:32] | type=2
  v8i g1;
  g1[0] = (int)(2u << 16);                      // data_size=4B; no mcast/iter/pad
  g1[1] = (int)((tile_w & 0xffffu) << 16);      // tensor_dim0[15:0]  (== tile_w)
  g1[2] = (int)(((tile_w >> 16) & 0xffffu) | ((tile_h & 0xffffu) << 16));
  g1[3] = (int)(((tile_h >> 16) & 0xffffu) | ((tile_w & 0xffffu) << 16)); // tile_dim0
  g1[4] = (int)(tile_h & 0xffffu);              // tile_dim1 (tile_dim2=0)
  g1[5] = (int)row_stride_elems;                // tensor_dim0_stride[31:0]
  g1[6] = 0;                                    // stride[47:32], dim1_stride lo
  g1[7] = 0;
  v4i z4 = {0, 0, 0, 0};
  v8i z8 = {0, 0, 0, 0, 0, 0, 0, 0};
  __builtin_amdgcn_tensor_load_to_lds(g0, g1, z4, z4, z8, 0);
}

// ---------------------------------------------------------------------------
// RMSNorm: one block per token, bf16 output for WMMA A-operands
// ---------------------------------------------------------------------------
__global__ __launch_bounds__(256) void rmsnorm_kernel(const float* __restrict__ x,
                                                      const float* __restrict__ w,
                                                      bf16_t* __restrict__ out, int D) {
  const int t = blockIdx.x;
  const float* xr = x + (size_t)t * D;
  __shared__ float red[8];
  __shared__ float sscale;
  float ss = 0.f;
  for (int i = threadIdx.x; i < D; i += blockDim.x) { float v = xr[i]; ss += v * v; }
#pragma unroll
  for (int off = 16; off; off >>= 1) ss += __shfl_xor(ss, off, 32);
  if ((threadIdx.x & 31) == 0) red[threadIdx.x >> 5] = ss;
  __syncthreads();
  if (threadIdx.x == 0) {
    float tot = 0.f;
    for (int i = 0; i < 8; ++i) tot += red[i];
    sscale = rsqrtf(tot / (float)D + RMS_EPS);
  }
  __syncthreads();
  const float sc = sscale;
  for (int i = threadIdx.x; i < D; i += blockDim.x)
    out[(size_t)t * D + i] = (bf16_t)(xr[i] * sc * w[i]);
}

// ---------------------------------------------------------------------------
// Generic GEMM: C[M,N] = A(bf16)[M,K] @ B(f32)[K,N], B tiles staged to LDS by
// the Tensor Data Mover (double-buffered, 1 outstanding op overlaps WMMAs).
// Block: 256 thr = 8 waves, tile 128x64; wave tile 32x32 (2x2 WMMA tiles).
// mode 0: f32 store; 1: bf16 store; 2: resid + alpha*acc;
// mode 3: resid + alpha*(acc + extra)   (final residual + shared + moe)
// ---------------------------------------------------------------------------
__device__ __forceinline__ void store_tile(v8f acc, int r0, int c, int ldc, void* Cout,
                                           const float* resid, const float* extra,
                                           float alpha, int mode, int half) {
#pragma unroll
  for (int vg = 0; vg < 8; ++vg) {
    int r = r0 + vg + 8 * half;
    size_t idx = (size_t)r * ldc + c;
    float v = acc[vg];
    if (mode == 0)      ((float*)Cout)[idx]  = v;
    else if (mode == 1) ((bf16_t*)Cout)[idx] = (bf16_t)v;
    else if (mode == 2) ((float*)Cout)[idx]  = resid[idx] + alpha * v;
    else                ((float*)Cout)[idx]  = resid[idx] + alpha * (v + extra[idx]);
  }
}

__global__ __launch_bounds__(256) void gemm_bf16_kernel(
    const bf16_t* __restrict__ A, int lda, const float* __restrict__ B, int ldb,
    void* __restrict__ Cout, int ldc, int K,
    const float* __restrict__ resid, const float* __restrict__ extra,
    float alpha, int mode) {
  const int lane = threadIdx.x & 31;
  const int wave = threadIdx.x >> 5;
  const int half = lane >> 4;
  const int ln   = lane & 15;
  const int m0 = blockIdx.y * 128 + (wave & 3) * 32;
  const int nblk = blockIdx.x * 64;
  const int nloc = (wave >> 2) * 32;
  v8f acc00 = {}, acc01 = {}, acc10 = {}, acc11 = {};
  const bf16_t* Ar0 = A + (size_t)(m0 + ln) * lda;
  const bf16_t* Ar1 = A + (size_t)(m0 + 16 + ln) * lda;

  __shared__ float Bs[2][32 * 64];   // double-buffered TDM destination (16 KB)
  const int nk = K / 32;
  if (wave == 0)
    tdm_load_2d_f32((unsigned)(unsigned long long)(void*)&Bs[0][0],
                    B + nblk, 64, 32, (unsigned)ldb);
  for (int kt = 0; kt < nk; ++kt) {
    const int k0 = kt * 32;
    if (kt + 1 < nk) {
      if (wave == 0) {
        tdm_load_2d_f32((unsigned)(unsigned long long)(void*)&Bs[(kt + 1) & 1][0],
                        B + (size_t)(k0 + 32) * ldb + nblk, 64, 32, (unsigned)ldb);
        __builtin_amdgcn_s_wait_tensorcnt(1);  // current tile landed in LDS
      }
    } else {
      if (wave == 0) __builtin_amdgcn_s_wait_tensorcnt(0);
    }
    __syncthreads();
    const float* Bt = &Bs[kt & 1][0];
    v16bf a0 = load_a_frag(Ar0, k0, half);
    v16bf a1 = load_a_frag(Ar1, k0, half);
    v16bf b0 = load_b_frag_lds(Bt, nloc + ln, half);
    v16bf b1 = load_b_frag_lds(Bt, nloc + 16 + ln, half);
    acc00 = wmma_bf16(a0, b0, acc00);
    acc01 = wmma_bf16(a0, b1, acc01);
    acc10 = wmma_bf16(a1, b0, acc10);
    acc11 = wmma_bf16(a1, b1, acc11);
    __syncthreads();  // all waves done with Bt before it is overwritten
  }
  const int n0 = nblk + nloc;
  store_tile(acc00, m0,      n0 + ln,      ldc, Cout, resid, extra, alpha, mode, half);
  store_tile(acc01, m0,      n0 + 16 + ln, ldc, Cout, resid, extra, alpha, mode, half);
  store_tile(acc10, m0 + 16, n0 + ln,      ldc, Cout, resid, extra, alpha, mode, half);
  store_tile(acc11, m0 + 16, n0 + 16 + ln, ldc, Cout, resid, extra, alpha, mode, half);
}

// ---------------------------------------------------------------------------
// RoPE (NeoX-style halves), emits bf16 Q/K for the attention WMMAs
// ---------------------------------------------------------------------------
__global__ __launch_bounds__(256) void rope_kernel(const float* __restrict__ qf,
                                                   const float* __restrict__ kf,
                                                   const int* __restrict__ pos,
                                                   bf16_t* __restrict__ qo,
                                                   bf16_t* __restrict__ ko) {
  const int t = blockIdx.x;
  const float p = (float)pos[t];
  for (int i = threadIdx.x; i < NH * 32; i += blockDim.x) {
    int h = i >> 5, d = i & 31;
    float ang = p * __powf(10000.f, -(float)d * (1.f / 32.f));
    float c = __cosf(ang), s = __sinf(ang);
    size_t o = (size_t)t * NH * HDIM + (size_t)h * HDIM + d;
    float x1 = qf[o], x2 = qf[o + 32];
    qo[o]      = (bf16_t)(x1 * c - x2 * s);
    qo[o + 32] = (bf16_t)(x2 * c + x1 * s);
  }
  for (int i = threadIdx.x; i < NKV * 32; i += blockDim.x) {
    int h = i >> 5, d = i & 31;
    float ang = p * __powf(10000.f, -(float)d * (1.f / 32.f));
    float c = __cosf(ang), s = __sinf(ang);
    size_t o = (size_t)t * NKV * HDIM + (size_t)h * HDIM + d;
    float x1 = kf[o], x2 = kf[o + 32];
    ko[o]      = (bf16_t)(x1 * c - x2 * s);
    ko[o + 32] = (bf16_t)(x2 * c + x1 * s);
  }
}

// ---------------------------------------------------------------------------
// Flash attention: 1 wave / (head, 16-query tile). Score tiles via WMMA
// (K=64 -> 2 steps), causal mask + online softmax with shfl_xor row
// reductions, P re-staged via 16x32 bf16 LDS to re-enter the A-fragment
// layout, P@V via 4 WMMAs.
// ---------------------------------------------------------------------------
__global__ __launch_bounds__(32) void flash_attn_kernel(
    const bf16_t* __restrict__ q, const bf16_t* __restrict__ k,
    const bf16_t* __restrict__ v, bf16_t* __restrict__ o) {
  const int lane = threadIdx.x;
  const int half = lane >> 4, ln = lane & 15;
  const int qbase = blockIdx.x * 16;
  const int h = blockIdx.y;
  const int kvh = h / (NH / NKV);
  const int qstride = NH * HDIM, kstride = NKV * HDIM;
  const bf16_t* Qr = q + (size_t)(qbase + ln) * qstride + h * HDIM;
  v16bf aq0 = load_a_frag(Qr, 0, half);
  v16bf aq1 = load_a_frag(Qr, 32, half);
  v8f accv[4] = {};
  float mrow[8], lrow[8];
#pragma unroll
  for (int i = 0; i < 8; ++i) { mrow[i] = -3.0e38f; lrow[i] = 0.f; }
  __shared__ bf16_t P[16][32];
  const int npairs = (qbase + 16 + 31) >> 5;
  for (int kp = 0; kp < npairs; ++kp) {
    const int kb = kp * 32;
    const int key0 = kb + ln;
    const int key1 = kb + 16 + ln;
    const int ck0 = key0 < T_TOK ? key0 : T_TOK - 1;
    const int ck1 = key1 < T_TOK ? key1 : T_TOK - 1;
    const bf16_t* K0 = k + (size_t)ck0 * kstride + kvh * HDIM;
    const bf16_t* K1 = k + (size_t)ck1 * kstride + kvh * HDIM;
    v8f s0 = {}, s1 = {};
#pragma unroll
    for (int kk = 0; kk < 2; ++kk) {
      const int d0 = kk * 32 + 16 * half;
      v16bf b0, b1;
#pragma unroll
      for (int e2 = 0; e2 < 16; ++e2) { b0[e2] = K0[d0 + e2]; b1[e2] = K1[d0 + e2]; }
      v16bf aa = kk ? aq1 : aq0;
      s0 = wmma_bf16(aa, b0, s0);
      s1 = wmma_bf16(aa, b1, s1);
    }
#pragma unroll
    for (int vg = 0; vg < 8; ++vg) {
      const int qi = qbase + vg + 8 * half;
      float v0 = s0[vg] * AMULT_F; if (key0 > qi) v0 = -3.0e38f;
      float v1 = s1[vg] * AMULT_F; if (key1 > qi) v1 = -3.0e38f;
      float rm = fmaxf(v0, v1);
#pragma unroll
      for (int off = 8; off; off >>= 1) rm = fmaxf(rm, __shfl_xor(rm, off, 32));
      float mnew  = fmaxf(mrow[vg], rm);
      float scale = __expf(mrow[vg] - mnew);
      float p0 = __expf(v0 - mnew);
      float p1 = __expf(v1 - mnew);
      float rs = p0 + p1;
#pragma unroll
      for (int off = 8; off; off >>= 1) rs += __shfl_xor(rs, off, 32);
      lrow[vg] = lrow[vg] * scale + rs;
      mrow[vg] = mnew;
#pragma unroll
      for (int dt = 0; dt < 4; ++dt) accv[dt][vg] *= scale;
      const int prow = vg + 8 * half;
      P[prow][ln]      = (bf16_t)p0;
      P[prow][16 + ln] = (bf16_t)p1;
    }
    __syncthreads();
    v16bf ap = load_a_frag(&P[ln][0], 0, half);
#pragma unroll
    for (int dt = 0; dt < 4; ++dt) {
      v16bf bv;
#pragma unroll
      for (int e2 = 0; e2 < 16; ++e2) {
        int key = kb + 16 * half + e2;
        if (key >= T_TOK) key = T_TOK - 1;  // P==0 there, contribution vanishes
        bv[e2] = v[(size_t)key * kstride + kvh * HDIM + dt * 16 + ln];
      }
      accv[dt] = wmma_bf16(ap, bv, accv[dt]);
    }
    __syncthreads();
  }
#pragma unroll
  for (int vg = 0; vg < 8; ++vg) {
    const float inv = 1.f / lrow[vg];
    const int row = qbase + vg + 8 * half;
    bf16_t* orow = o + (size_t)row * qstride + h * HDIM + ln;
#pragma unroll
    for (int dt = 0; dt < 4; ++dt) orow[dt * 16] = (bf16_t)(accv[dt][vg] * inv);
  }
}

// ---------------------------------------------------------------------------
// Router: 32 logits/token (lane==expert), top-8 + softmax by lane 0
// ---------------------------------------------------------------------------
__global__ __launch_bounds__(32) void router_topk_kernel(
    const bf16_t* __restrict__ h2, const float* __restrict__ rw,
    int* __restrict__ topi, float* __restrict__ gates) {
  const int t = blockIdx.x;
  const int lane = threadIdx.x;
  __shared__ float lg[NEXP];
  const bf16_t* hr = h2 + (size_t)t * DMODEL;
  float acc = 0.f;
  for (int d = 0; d < DMODEL; ++d) acc += (float)hr[d] * rw[(size_t)d * NEXP + lane];
  lg[lane] = acc;
  __syncthreads();
  if (lane == 0) {
    float lv[NEXP];
    for (int i = 0; i < NEXP; ++i) lv[i] = lg[i];
    int idx[TOPK_N]; float val[TOPK_N];
    for (int j = 0; j < TOPK_N; ++j) {
      int bi = 0; float bv = -3.0e38f;
      for (int i = 0; i < NEXP; ++i) if (lv[i] > bv) { bv = lv[i]; bi = i; }
      idx[j] = bi; val[j] = bv; lv[bi] = -3.0e38f;
    }
    float mx = val[0], sum = 0.f, pr[TOPK_N];
    for (int j = 0; j < TOPK_N; ++j) { pr[j] = __expf(val[j] - mx); sum += pr[j]; }
    float inv = 1.f / sum;
    for (int j = 0; j < TOPK_N; ++j) {
      topi[(size_t)t * TOPK_N + j]  = idx[j];
      gates[(size_t)t * TOPK_N + j] = pr[j] * inv;
    }
  }
}

// ---- expert-list building (gathered MoE: 4x fewer FLOPs than dense) --------
__global__ void count_kernel(const int* __restrict__ topi, int* __restrict__ counts, int n) {
  int i = blockIdx.x * blockDim.x + threadIdx.x;
  if (i < n) atomicAdd(&counts[topi[i]], 1);
}
__global__ void scan_kernel(const int* __restrict__ counts, int* __restrict__ offsets) {
  if (threadIdx.x == 0) {
    int s = 0;
    for (int e = 0; e < NEXP; ++e) { offsets[e] = s; s += counts[e]; }
    offsets[NEXP] = s;
  }
}
__global__ void place_kernel(const int* __restrict__ topi, const float* __restrict__ gates,
                             const int* __restrict__ offsets, int* __restrict__ cursor,
                             int* __restrict__ etok, float* __restrict__ egate, int n) {
  int i = blockIdx.x * blockDim.x + threadIdx.x;
  if (i < n) {
    int e = topi[i];
    int slot = offsets[e] + atomicAdd(&cursor[e], 1);
    etok[slot]  = i >> 3;   // token = entry / TOPK
    egate[slot] = gates[i];
  }
}

// ---------------------------------------------------------------------------
// gate_up GEMM with gathered rows + fused SiLU(g)*u -> bf16 activations.
// offsets==nullptr => shared expert (identity rows, cnt=Mtotal).
// ---------------------------------------------------------------------------
__device__ __forceinline__ void store_act(v8f g, v8f u, int rbase, int c, int cnt,
                                          int base, int F, bf16_t* __restrict__ act,
                                          int half) {
#pragma unroll
  for (int vg = 0; vg < 8; ++vg) {
    int r = rbase + vg + 8 * half;
    if (r < cnt) {
      float gg = g[vg];
      float s = gg / (1.f + __expf(-gg));
      act[(size_t)(base + r) * F + c] = (bf16_t)(s * u[vg]);
    }
  }
}

__global__ __launch_bounds__(256) void gemm_gateup_kernel(
    const bf16_t* __restrict__ A, const float* __restrict__ Bbase,
    bf16_t* __restrict__ act, const int* __restrict__ entry_token,
    const int* __restrict__ offsets, int Mtotal, int F, int Dk, long long strideBe) {
  const int e = blockIdx.z;
  const int base = offsets ? offsets[e] : 0;
  const int cnt  = offsets ? (offsets[e + 1] - base) : Mtotal;
  const int lane = threadIdx.x & 31;
  const int wave = threadIdx.x >> 5;
  const int half = lane >> 4, ln = lane & 15;
  const int m0 = blockIdx.y * 128 + (wave & 3) * 32;
  if (m0 >= cnt) return;
  const int n0 = blockIdx.x * 64 + (wave >> 2) * 32;
  const float* B = Bbase + (size_t)e * strideBe;
  const int ldb = 2 * F;
  int r0 = m0 + ln;      if (r0 > cnt - 1) r0 = cnt - 1;
  int r1 = m0 + 16 + ln; if (r1 > cnt - 1) r1 = cnt - 1;
  const int tok0 = entry_token ? entry_token[base + r0] : r0;
  const int tok1 = entry_token ? entry_token[base + r1] : r1;
  const bf16_t* Ar0 = A + (size_t)tok0 * Dk;
  const bf16_t* Ar1 = A + (size_t)tok1 * Dk;
  v8f g00 = {}, g01 = {}, g10 = {}, g11 = {};
  v8f u00 = {}, u01 = {}, u10 = {}, u11 = {};
  for (int k0 = 0; k0 < Dk; k0 += 32) {
    v16bf a0 = load_a_frag(Ar0, k0, half);
    v16bf a1 = load_a_frag(Ar1, k0, half);
    v16bf bg0 = load_b_frag_f32(B, ldb, k0, n0 + ln, half);
    v16bf bg1 = load_b_frag_f32(B, ldb, k0, n0 + 16 + ln, half);
    v16bf bu0 = load_b_frag_f32(B, ldb, k0, F + n0 + ln, half);
    v16bf bu1 = load_b_frag_f32(B, ldb, k0, F + n0 + 16 + ln, half);
    g00 = wmma_bf16(a0, bg0, g00); g01 = wmma_bf16(a0, bg1, g01);
    g10 = wmma_bf16(a1, bg0, g10); g11 = wmma_bf16(a1, bg1, g11);
    u00 = wmma_bf16(a0, bu0, u00); u01 = wmma_bf16(a0, bu1, u01);
    u10 = wmma_bf16(a1, bu0, u10); u11 = wmma_bf16(a1, bu1, u11);
  }
  store_act(g00, u00, m0,      n0 + ln,      cnt, base, F, act, half);
  store_act(g01, u01, m0,      n0 + 16 + ln, cnt, base, F, act, half);
  store_act(g10, u10, m0 + 16, n0 + ln,      cnt, base, F, act, half);
  store_act(g11, u11, m0 + 16, n0 + 16 + ln, cnt, base, F, act, half);
}

// ---------------------------------------------------------------------------
// down GEMM: act(entries, F) @ w_down[e](F, D); gate-weighted atomic scatter
// into moe_out[token] (global_atomic_add_f32)
// ---------------------------------------------------------------------------
__device__ __forceinline__ void scatter_tile(v8f acc, int rbase, int c, int cnt, int base,
                                             const int* __restrict__ entry_token,
                                             const float* __restrict__ entry_gate,
                                             float* __restrict__ moe_out, int Dm, int half) {
#pragma unroll
  for (int vg = 0; vg < 8; ++vg) {
    int r = rbase + vg + 8 * half;
    if (r < cnt) {
      int tok = entry_token[base + r];
      float gate = entry_gate[base + r];
      atomicAdd(&moe_out[(size_t)tok * Dm + c], gate * acc[vg]);
    }
  }
}

__global__ __launch_bounds__(256) void gemm_down_kernel(
    const bf16_t* __restrict__ act, const float* __restrict__ Bbase,
    float* __restrict__ moe_out, const int* __restrict__ entry_token,
    const float* __restrict__ entry_gate, const int* __restrict__ offsets,
    int F, int Dm, long long strideBe) {
  const int e = blockIdx.z;
  const int base = offsets[e];
  const int cnt  = offsets[e + 1] - base;
  const int lane = threadIdx.x & 31;
  const int wave = threadIdx.x >> 5;
  const int half = lane >> 4, ln = lane & 15;
  const int m0 = blockIdx.y * 128 + (wave & 3) * 32;
  if (m0 >= cnt) return;
  const int n0 = blockIdx.x * 64 + (wave >> 2) * 32;
  const float* B = Bbase + (size_t)e * strideBe;
  int r0 = m0 + ln;      if (r0 > cnt - 1) r0 = cnt - 1;
  int r1 = m0 + 16 + ln; if (r1 > cnt - 1) r1 = cnt - 1;
  const bf16_t* Ar0 = act + (size_t)(base + r0) * F;
  const bf16_t* Ar1 = act + (size_t)(base + r1) * F;
  v8f a00 = {}, a01 = {}, a10 = {}, a11 = {};
  for (int k0 = 0; k0 < F; k0 += 32) {
    v16bf f0 = load_a_frag(Ar0, k0, half);
    v16bf f1 = load_a_frag(Ar1, k0, half);
    v16bf b0 = load_b_frag_f32(B, Dm, k0, n0 + ln, half);
    v16bf b1 = load_b_frag_f32(B, Dm, k0, n0 + 16 + ln, half);
    a00 = wmma_bf16(f0, b0, a00); a01 = wmma_bf16(f0, b1, a01);
    a10 = wmma_bf16(f1, b0, a10); a11 = wmma_bf16(f1, b1, a11);
  }
  scatter_tile(a00, m0,      n0 + ln,      cnt, base, entry_token, entry_gate, moe_out, Dm, half);
  scatter_tile(a01, m0,      n0 + 16 + ln, cnt, base, entry_token, entry_gate, moe_out, Dm, half);
  scatter_tile(a10, m0 + 16, n0 + ln,      cnt, base, entry_token, entry_gate, moe_out, Dm, half);
  scatter_tile(a11, m0 + 16, n0 + 16 + ln, cnt, base, entry_token, entry_gate, moe_out, Dm, half);
}

// ---------------------------------------------------------------------------
extern "C" void kernel_launch(void* const* d_in, const int* in_sizes, int n_in,
                              void* d_out, int out_size, void* d_ws, size_t ws_size,
                              hipStream_t stream) {
  (void)in_sizes; (void)n_in; (void)out_size; (void)ws_size;
  const int*   positions = (const int*)d_in[0];
  const float* hidden    = (const float*)d_in[1];
  const float* ln1_w     = (const float*)d_in[2];
  const float* wq        = (const float*)d_in[3];
  const float* wk        = (const float*)d_in[4];
  const float* wv        = (const float*)d_in[5];
  const float* wo        = (const float*)d_in[6];
  const float* ln2_w     = (const float*)d_in[7];
  const float* router_w  = (const float*)d_in[8];
  const float* w_gate_up = (const float*)d_in[9];
  const float* w_down    = (const float*)d_in[10];
  const float* sh_gu     = (const float*)d_in[11];
  const float* sh_down   = (const float*)d_in[12];
  float* out = (float*)d_out;

  // workspace carve (~185 MB total)
  char* p = (char*)d_ws;
  auto alloc = [&](size_t bytes) -> void* {
    void* q = (void*)p;
    p += (bytes + 255) & ~(size_t)255;
    return q;
  };
  bf16_t* h1      = (bf16_t*)alloc((size_t)T_TOK * DMODEL * 2);
  float*  qf      = (float*)alloc((size_t)T_TOK * NH * HDIM * 4);
  float*  kf      = (float*)alloc((size_t)T_TOK * NKV * HDIM * 4);
  bf16_t* qb      = (bf16_t*)alloc((size_t)T_TOK * NH * HDIM * 2);
  bf16_t* kb      = (bf16_t*)alloc((size_t)T_TOK * NKV * HDIM * 2);
  bf16_t* vb      = (bf16_t*)alloc((size_t)T_TOK * NKV * HDIM * 2);
  bf16_t* attnb   = (bf16_t*)alloc((size_t)T_TOK * NH * HDIM * 2);
  float*  hidden1 = (float*)alloc((size_t)T_TOK * DMODEL * 4);
  bf16_t* h2      = (bf16_t*)alloc((size_t)T_TOK * DMODEL * 2);
  int*    topi    = (int*)alloc((size_t)T_TOK * TOPK_N * 4);
  float*  gates   = (float*)alloc((size_t)T_TOK * TOPK_N * 4);
  int*    counts  = (int*)alloc(NEXP * 4);
  int*    offsets = (int*)alloc((NEXP + 1) * 4);
  int*    cursor  = (int*)alloc(NEXP * 4);
  int*    etok    = (int*)alloc((size_t)T_TOK * TOPK_N * 4);
  float*  egate   = (float*)alloc((size_t)T_TOK * TOPK_N * 4);
  bf16_t* act_moe = (bf16_t*)alloc((size_t)T_TOK * TOPK_N * FEXP * 2);
  bf16_t* act_sh  = (bf16_t*)alloc((size_t)T_TOK * FSHARED * 2);
  float*  moe_out = (float*)alloc((size_t)T_TOK * DMODEL * 4);

  (void)hipMemsetAsync(counts, 0, NEXP * 4, stream);
  (void)hipMemsetAsync(cursor, 0, NEXP * 4, stream);
  (void)hipMemsetAsync(moe_out, 0, (size_t)T_TOK * DMODEL * 4, stream);

  const dim3 blk(256);
  // --- attention branch ---
  rmsnorm_kernel<<<T_TOK, 256, 0, stream>>>(hidden, ln1_w, h1, DMODEL);
  gemm_bf16_kernel<<<dim3(NH * HDIM / 64, T_TOK / 128), blk, 0, stream>>>(
      h1, DMODEL, wq, NH * HDIM, qf, NH * HDIM, DMODEL, nullptr, nullptr, 0.f, 0);
  gemm_bf16_kernel<<<dim3(NKV * HDIM / 64, T_TOK / 128), blk, 0, stream>>>(
      h1, DMODEL, wk, NKV * HDIM, kf, NKV * HDIM, DMODEL, nullptr, nullptr, 0.f, 0);
  gemm_bf16_kernel<<<dim3(NKV * HDIM / 64, T_TOK / 128), blk, 0, stream>>>(
      h1, DMODEL, wv, NKV * HDIM, vb, NKV * HDIM, DMODEL, nullptr, nullptr, 0.f, 1);
  rope_kernel<<<T_TOK, 256, 0, stream>>>(qf, kf, positions, qb, kb);
  flash_attn_kernel<<<dim3(T_TOK / 16, NH), 32, 0, stream>>>(qb, kb, vb, attnb);
  gemm_bf16_kernel<<<dim3(DMODEL / 64, T_TOK / 128), blk, 0, stream>>>(
      attnb, NH * HDIM, wo, DMODEL, hidden1, DMODEL, NH * HDIM,
      hidden, nullptr, RMULT_F, 2);
  // --- MoE branch ---
  rmsnorm_kernel<<<T_TOK, 256, 0, stream>>>(hidden1, ln2_w, h2, DMODEL);
  router_topk_kernel<<<T_TOK, 32, 0, stream>>>(h2, router_w, topi, gates);
  count_kernel<<<(T_TOK * TOPK_N + 255) / 256, 256, 0, stream>>>(topi, counts, T_TOK * TOPK_N);
  scan_kernel<<<1, 1, 0, stream>>>(counts, offsets);
  place_kernel<<<(T_TOK * TOPK_N + 255) / 256, 256, 0, stream>>>(
      topi, gates, offsets, cursor, etok, egate, T_TOK * TOPK_N);
  gemm_gateup_kernel<<<dim3(FEXP / 64, T_TOK / 128, NEXP), blk, 0, stream>>>(
      h2, w_gate_up, act_moe, etok, offsets, 0, FEXP, DMODEL,
      (long long)DMODEL * 2 * FEXP);
  gemm_gateup_kernel<<<dim3(FSHARED / 64, T_TOK / 128, 1), blk, 0, stream>>>(
      h2, sh_gu, act_sh, nullptr, nullptr, T_TOK, FSHARED, DMODEL, 0);
  gemm_down_kernel<<<dim3(DMODEL / 64, T_TOK / 128, NEXP), blk, 0, stream>>>(
      act_moe, w_down, moe_out, etok, egate, offsets, FEXP, DMODEL,
      (long long)FEXP * DMODEL);
  // final: out = hidden1 + RMULT * (shared_down(act_sh) + moe_out)
  gemm_bf16_kernel<<<dim3(DMODEL / 64, T_TOK / 128), blk, 0, stream>>>(
      act_sh, FSHARED, sh_down, DMODEL, out, DMODEL, FSHARED,
      hidden1, moe_out, RMULT_F, 3);
}
